// TritonLinearModule_30949534335490
// MI455X (gfx1250) — matrix operand order repaired and checked by reading, not compiled, not run on hardware
//
#include <hip/hip_runtime.h>
#include <hip/hip_bf16.h>
#include <cstdint>

// ---------------------------------------------------------------------------
// Screened linear layer for MI455X (gfx1250, wave32, WMMA).
//   out = (tstat < TAU ? 0 : x@W^T) + bias, tstat from first K0=256 columns.
// v3: fix async-builtin arg types (int4 vector pointees, non-const).
// fp32->bf16 hi/lo split preprocessed once into workspace; compute kernel hot
// loop = global_load_async_to_lds (ASYNCcnt) + ds_load fragments + WMMA.
// ---------------------------------------------------------------------------

typedef __attribute__((ext_vector_type(16))) __bf16 v16bf;
typedef __attribute__((ext_vector_type(8)))  __bf16 v8bf;
typedef __attribute__((ext_vector_type(4)))  __bf16 v4bf;
typedef __attribute__((ext_vector_type(8)))  float  v8f;
typedef int v4i_raw __attribute__((vector_size(16)));   // matches builtin pointee

#define M_DIM 8192
#define N_DIM 4096
#define K_DIM 4096
#define BM 128
#define BN 128
#define BK 32
#define LDK 40                   // padded LDS row stride (bf16): 80B, 16B-aligned, 20-bank stride
#define NKT (K_DIM / BK)         // 128 K-steps
#define KT0 (256 / BK)           // screening region = first 8 K-steps
#define TILE_ELEMS (BM * LDK)    // 5120 bf16 per LDS sub-array

#define AS1 __attribute__((address_space(1)))
#define AS3 __attribute__((address_space(3)))

// ---- CDNA5 async global->LDS copy (16B per lane) ---------------------------
#if __has_builtin(__builtin_amdgcn_global_load_async_to_lds_b128)
#define ASYNC_CP16(gp, lp)                                                     \
    __builtin_amdgcn_global_load_async_to_lds_b128(                            \
        (AS1 v4i_raw*)(uintptr_t)(gp), (AS3 v4i_raw*)(uintptr_t)(lp), 0, 0)
#else
#define ASYNC_CP16(gp, lp)                                                     \
    asm volatile("global_load_async_to_lds_b128 %0, %1, off"                   \
                 :: "v"((AS3 char*)(uintptr_t)(lp)),                           \
                    "v"((AS1 char*)(uintptr_t)(gp))                            \
                 : "memory")
#endif

#if __has_builtin(__builtin_amdgcn_s_wait_asynccnt)
#define WAIT_ASYNC(n) __builtin_amdgcn_s_wait_asynccnt(n)
#else
#define WAIT_ASYNC(n) asm volatile("s_wait_asynccnt %0" :: "n"(n) : "memory")
#endif

__device__ __forceinline__ v16bf ld_frag(const __bf16* p0, const __bf16* p1) {
    v8bf a = *(const v8bf*)p0;
    v8bf b = *(const v8bf*)p1;
    return __builtin_shufflevector(a, b, 0,1,2,3,4,5,6,7,8,9,10,11,12,13,14,15);
}

__device__ __forceinline__ v16bf square_frag(v16bf h, v16bf l) {
    v16bf r;
#pragma unroll
    for (int i = 0; i < 16; ++i) {
        float f = (float)h[i] + (float)l[i];
        r[i] = (__bf16)(f * f);
    }
    return r;
}

// ---------------------------------------------------------------------------
// Preprocess: split fp32 -> bf16 (hi) + bf16 (lo = RNE(f - hi)). Pure BW pass.
// ---------------------------------------------------------------------------
__global__ __launch_bounds__(256)
void split_bf16_kernel(const float* __restrict__ in,
                       __bf16* __restrict__ hi, __bf16* __restrict__ lo, int n4) {
    const int i = blockIdx.x * 256 + threadIdx.x;
    if (i >= n4) return;
    float4 v = ((const float4*)in)[i];
    const float f[4] = {v.x, v.y, v.z, v.w};
    v4bf h, l;
#pragma unroll
    for (int j = 0; j < 4; ++j) {
        __bf16 hh = (__bf16)f[j];
        h[j] = hh;
        l[j] = (__bf16)(f[j] - (float)hh);
    }
    ((v4bf*)hi)[i] = h;
    ((v4bf*)lo)[i] = l;
}

// ---------------------------------------------------------------------------
// Fast path: bf16 hi/lo split-GEMM, async-copy staged, double-buffered LDS.
// ---------------------------------------------------------------------------
__global__ __launch_bounds__(256)
void screened_linear_async(const __bf16* __restrict__ xhi, const __bf16* __restrict__ xlo,
                           const __bf16* __restrict__ whi, const __bf16* __restrict__ wlo,
                           const float* __restrict__ bias, float* __restrict__ out) {
    // [buf][arr][elem]; arr: 0=Ahi 1=Alo 2=Bhi 3=Blo. 80KB total (<320KB/WGP).
    __shared__ __bf16 SH[2][4][TILE_ELEMS];

    const int t     = threadIdx.x;
    const int lane  = t & 31;
    const int wid   = t >> 5;        // 0..7
    const int waveM = wid & 3;       // 4 waves x 32 rows
    const int waveN = wid >> 2;      // 2 waves x 64 cols
    const int row0  = blockIdx.y * BM;
    const int col0  = blockIdx.x * BN;

    v8f acc[2][4];
    v8f sacc[2][4];
    const v8f vzero = {0.f, 0.f, 0.f, 0.f, 0.f, 0.f, 0.f, 0.f};
#pragma unroll
    for (int mt = 0; mt < 2; ++mt)
#pragma unroll
        for (int nt = 0; nt < 4; ++nt) { acc[mt][nt] = vzero; sacc[mt][nt] = vzero; }

    // Staging: a BMxBK bf16 tile is 512 16B-chunks; 2 chunks per thread per array.
    // Chunk c: row = c>>2, 16B-slot q = c&3 (8 bf16 elems each).
    auto issue = [&](int buf, int kt) {
#pragma unroll
        for (int j = 0; j < 2; ++j) {
            const int c = t * 2 + j;
            const int r = c >> 2;
            const int q = (c & 3) * 8;
            const size_t gx = (size_t)(row0 + r) * K_DIM + (size_t)kt * BK + q;
            const size_t gw = (size_t)(col0 + r) * K_DIM + (size_t)kt * BK + q;
            const int ls = r * LDK + q;
            ASYNC_CP16(xhi + gx, &SH[buf][0][ls]);
            ASYNC_CP16(xlo + gx, &SH[buf][1][ls]);
            ASYNC_CP16(whi + gw, &SH[buf][2][ls]);
            ASYNC_CP16(wlo + gw, &SH[buf][3][ls]);
        }
    };

    unsigned mask[2][4] = {{0u, 0u, 0u, 0u}, {0u, 0u, 0u, 0u}};

    issue(0, 0);                     // prologue: 8 async ops in flight

    for (int kt = 0; kt < NKT; ++kt) {
        const int cur = kt & 1;
        if (kt + 1 < NKT) {
            issue(cur ^ 1, kt + 1);  // prefetch next tile (8 more in flight)
            WAIT_ASYNC(8);           // previous 8 (current tile) have landed
        } else {
            WAIT_ASYNC(0);
        }
        __syncthreads();             // all waves' current tile visible

        // --- fragments (ISA 7.12.2 layouts) ---
        const int aoff = (lane >> 4) ? 8 : 0;    // A: lane<16 -> K 0-7 & 16-23
        const int boff = (lane >> 4) ? 8 : 0;    // B: lane<16 -> K 0-7 & 16-23
        v16bf ahi[2], alo[2];
#pragma unroll
        for (int mt = 0; mt < 2; ++mt) {
            const int m = waveM * 32 + mt * 16 + (lane & 15);
            const __bf16* ph = &SH[cur][0][m * LDK + aoff];
            const __bf16* pl = &SH[cur][1][m * LDK + aoff];
            ahi[mt] = ld_frag(ph, ph + 16);
            alo[mt] = ld_frag(pl, pl + 16);
        }
        v16bf bhi[4], blo[4];
#pragma unroll
        for (int nt = 0; nt < 4; ++nt) {
            const int n = waveN * 64 + nt * 16 + (lane & 15);
            const __bf16* ph = &SH[cur][2][n * LDK + boff];
            const __bf16* pl = &SH[cur][3][n * LDK + boff];
            bhi[nt] = ld_frag(ph, ph + 16);
            blo[nt] = ld_frag(pl, pl + 16);
        }

        // --- split-precision GEMM: hi*hi + hi*lo + lo*hi ---
#pragma unroll
        for (int mt = 0; mt < 2; ++mt)
#pragma unroll
            for (int nt = 0; nt < 4; ++nt) {
                acc[mt][nt] = __builtin_amdgcn_wmma_f32_16x16x32_bf16(
                    false, ahi[mt], false, bhi[nt], (short)0, acc[mt][nt], false, false);
                acc[mt][nt] = __builtin_amdgcn_wmma_f32_16x16x32_bf16(
                    false, ahi[mt], false, blo[nt], (short)0, acc[mt][nt], false, false);
                acc[mt][nt] = __builtin_amdgcn_wmma_f32_16x16x32_bf16(
                    false, alo[mt], false, bhi[nt], (short)0, acc[mt][nt], false, false);
            }

        // --- screening region: s2 += (x^2)(w^2) via squared fragments ---
        if (kt < KT0) {
            v16bf asq[2], bsq[4];
#pragma unroll
            for (int mt = 0; mt < 2; ++mt) asq[mt] = square_frag(ahi[mt], alo[mt]);
#pragma unroll
            for (int nt = 0; nt < 4; ++nt) bsq[nt] = square_frag(bhi[nt], blo[nt]);
#pragma unroll
            for (int mt = 0; mt < 2; ++mt)
#pragma unroll
                for (int nt = 0; nt < 4; ++nt)
                    sacc[mt][nt] = __builtin_amdgcn_wmma_f32_16x16x32_bf16(
                        false, asq[mt], false, bsq[nt], (short)0, sacc[mt][nt], false, false);
        }

        // --- at k==256: acc==y1, sacc==s2 -> pass mask ---
        if (kt == KT0 - 1) {
#pragma unroll
            for (int mt = 0; mt < 2; ++mt)
#pragma unroll
                for (int nt = 0; nt < 4; ++nt) {
                    unsigned mb = 0u;
#pragma unroll
                    for (int i = 0; i < 8; ++i) {
                        const float y1 = acc[mt][nt][i];
                        const float s2 = sacc[mt][nt][i];
                        float denom = __builtin_sqrtf(s2 * (1.0f / 256.0f) + 1e-6f);
                        denom = fmaxf(denom, 1e-6f);
                        if (fabsf(y1) < 3.0f * denom) mb |= (1u << i);
                    }
                    mask[mt][nt] = mb;
                }
        }

        __syncthreads();             // everyone done reading; next buffer reusable
    }

    // --- epilogue: mask + bias; C layout: vgpr i -> M=i (lanes<16) / i+8 ---
#pragma unroll
    for (int mt = 0; mt < 2; ++mt) {
        const int mbase = row0 + waveM * 32 + mt * 16 + ((lane >> 4) ? 8 : 0);
#pragma unroll
        for (int nt = 0; nt < 4; ++nt) {
            const int n = col0 + waveN * 64 + nt * 16 + (lane & 15);
            const float bv = bias[n];
            const unsigned mb = mask[mt][nt];
#pragma unroll
            for (int i = 0; i < 8; ++i) {
                const float v = ((mb >> i) & 1u) ? 0.0f : acc[mt][nt][i];
                out[(size_t)(mbase + i) * N_DIM + n] = v + bv;
            }
        }
    }
}

// ---------------------------------------------------------------------------
// Fallback (no workspace): v1-style in-kernel conversion. Correct, slower.
// ---------------------------------------------------------------------------
__global__ __launch_bounds__(256)
void screened_linear_fallback(const float* __restrict__ x,
                              const float* __restrict__ w,
                              const float* __restrict__ bias,
                              float* __restrict__ out) {
    __shared__ __bf16 Ahi[2][TILE_ELEMS];
    __shared__ __bf16 Alo[2][TILE_ELEMS];
    __shared__ __bf16 Bhi[2][TILE_ELEMS];
    __shared__ __bf16 Blo[2][TILE_ELEMS];

    const int t     = threadIdx.x;
    const int lane  = t & 31;
    const int wid   = t >> 5;
    const int waveM = wid & 3;
    const int waveN = wid >> 2;
    const int row0  = blockIdx.y * BM;
    const int col0  = blockIdx.x * BN;
    const int lr    = t >> 3;
    const int lc    = (t & 7) * 4;

    v8f acc[2][4];
    v8f sacc[2][4];
    const v8f vzero = {0.f, 0.f, 0.f, 0.f, 0.f, 0.f, 0.f, 0.f};
#pragma unroll
    for (int mt = 0; mt < 2; ++mt)
#pragma unroll
        for (int nt = 0; nt < 4; ++nt) { acc[mt][nt] = vzero; sacc[mt][nt] = vzero; }

    float4 xr[4], wr[4];
    auto gload = [&](int kt) {
        const int k0 = kt * BK;
#pragma unroll
        for (int p = 0; p < 4; ++p) {
            xr[p] = *(const float4*)&x[(size_t)(row0 + p * 32 + lr) * K_DIM + k0 + lc];
            wr[p] = *(const float4*)&w[(size_t)(col0 + p * 32 + lr) * K_DIM + k0 + lc];
        }
    };
    auto lstore = [&](int buf) {
#pragma unroll
        for (int p = 0; p < 4; ++p) {
            const int ra = p * 32 + lr;
            float fx[4] = {xr[p].x, xr[p].y, xr[p].z, xr[p].w};
            v4bf hi, lo;
#pragma unroll
            for (int i = 0; i < 4; ++i) {
                __bf16 h = (__bf16)fx[i]; hi[i] = h; lo[i] = (__bf16)(fx[i] - (float)h);
            }
            *(v4bf*)&Ahi[buf][ra * LDK + lc] = hi;
            *(v4bf*)&Alo[buf][ra * LDK + lc] = lo;
            float fw[4] = {wr[p].x, wr[p].y, wr[p].z, wr[p].w};
#pragma unroll
            for (int i = 0; i < 4; ++i) {
                __bf16 h = (__bf16)fw[i]; hi[i] = h; lo[i] = (__bf16)(fw[i] - (float)h);
            }
            *(v4bf*)&Bhi[buf][ra * LDK + lc] = hi;
            *(v4bf*)&Blo[buf][ra * LDK + lc] = lo;
        }
    };

    unsigned mask[2][4] = {{0u, 0u, 0u, 0u}, {0u, 0u, 0u, 0u}};
    gload(0);
    lstore(0);
    __syncthreads();

    for (int kt = 0; kt < NKT; ++kt) {
        const int cur = kt & 1;
        if (kt + 1 < NKT) gload(kt + 1);

        const int aoff = (lane >> 4) ? 8 : 0;
        const int boff = (lane >> 4) ? 8 : 0;
        v16bf ahi[2], alo[2];
#pragma unroll
        for (int mt = 0; mt < 2; ++mt) {
            const int m = waveM * 32 + mt * 16 + (lane & 15);
            const __bf16* ph = &Ahi[cur][m * LDK + aoff];
            const __bf16* pl = &Alo[cur][m * LDK + aoff];
            ahi[mt] = ld_frag(ph, ph + 16);
            alo[mt] = ld_frag(pl, pl + 16);
        }
        v16bf bhi[4], blo[4];
#pragma unroll
        for (int nt = 0; nt < 4; ++nt) {
            const int n = waveN * 64 + nt * 16 + (lane & 15);
            const __bf16* ph = &Bhi[cur][n * LDK + boff];
            const __bf16* pl = &Blo[cur][n * LDK + boff];
            bhi[nt] = ld_frag(ph, ph + 16);
            blo[nt] = ld_frag(pl, pl + 16);
        }

#pragma unroll
        for (int mt = 0; mt < 2; ++mt)
#pragma unroll
            for (int nt = 0; nt < 4; ++nt) {
                acc[mt][nt] = __builtin_amdgcn_wmma_f32_16x16x32_bf16(
                    false, ahi[mt], false, bhi[nt], (short)0, acc[mt][nt], false, false);
                acc[mt][nt] = __builtin_amdgcn_wmma_f32_16x16x32_bf16(
                    false, ahi[mt], false, blo[nt], (short)0, acc[mt][nt], false, false);
                acc[mt][nt] = __builtin_amdgcn_wmma_f32_16x16x32_bf16(
                    false, alo[mt], false, bhi[nt], (short)0, acc[mt][nt], false, false);
            }

        if (kt < KT0) {
            v16bf asq[2], bsq[4];
#pragma unroll
            for (int mt = 0; mt < 2; ++mt) asq[mt] = square_frag(ahi[mt], alo[mt]);
#pragma unroll
            for (int nt = 0; nt < 4; ++nt) bsq[nt] = square_frag(bhi[nt], blo[nt]);
#pragma unroll
            for (int mt = 0; mt < 2; ++mt)
#pragma unroll
                for (int nt = 0; nt < 4; ++nt)
                    sacc[mt][nt] = __builtin_amdgcn_wmma_f32_16x16x32_bf16(
                        false, asq[mt], false, bsq[nt], (short)0, sacc[mt][nt], false, false);
        }

        if (kt == KT0 - 1) {
#pragma unroll
            for (int mt = 0; mt < 2; ++mt)
#pragma unroll
                for (int nt = 0; nt < 4; ++nt) {
                    unsigned mb = 0u;
#pragma unroll
                    for (int i = 0; i < 8; ++i) {
                        const float y1 = acc[mt][nt][i];
                        const float s2 = sacc[mt][nt][i];
                        float denom = __builtin_sqrtf(s2 * (1.0f / 256.0f) + 1e-6f);
                        denom = fmaxf(denom, 1e-6f);
                        if (fabsf(y1) < 3.0f * denom) mb |= (1u << i);
                    }
                    mask[mt][nt] = mb;
                }
        }

        if (kt + 1 < NKT) lstore(cur ^ 1);
        __syncthreads();
    }

#pragma unroll
    for (int mt = 0; mt < 2; ++mt) {
        const int mbase = row0 + waveM * 32 + mt * 16 + ((lane >> 4) ? 8 : 0);
#pragma unroll
        for (int nt = 0; nt < 4; ++nt) {
            const int n = col0 + waveN * 64 + nt * 16 + (lane & 15);
            const float bv = bias[n];
            const unsigned mb = mask[mt][nt];
#pragma unroll
            for (int i = 0; i < 8; ++i) {
                const float v = ((mb >> i) & 1u) ? 0.0f : acc[mt][nt][i];
                out[(size_t)(mbase + i) * N_DIM + n] = v + bv;
            }
        }
    }
}

extern "C" void kernel_launch(void* const* d_in, const int* in_sizes, int n_in,
                              void* d_out, int out_size, void* d_ws, size_t ws_size,
                              hipStream_t stream) {
    (void)in_sizes; (void)n_in; (void)out_size;
    const float* x    = (const float*)d_in[0];
    const float* w    = (const float*)d_in[1];
    const float* bias = (const float*)d_in[2];
    float* out        = (float*)d_out;

    const size_t xElems = (size_t)M_DIM * K_DIM;   // 33.5M
    const size_t wElems = (size_t)N_DIM * K_DIM;   // 16.8M
    const size_t need   = (2 * xElems + 2 * wElems) * sizeof(__bf16);  // ~192 MiB

    dim3 grid(N_DIM / BN, M_DIM / BM);             // (32, 64), 256 thr = 8 wave32

    if (ws_size >= need) {
        __bf16* xhi = (__bf16*)d_ws;
        __bf16* xlo = xhi + xElems;
        __bf16* whi = xlo + xElems;
        __bf16* wlo = whi + wElems;
        const int xq = (int)(xElems / 4), wq = (int)(wElems / 4);
        split_bf16_kernel<<<(xq + 255) / 256, 256, 0, stream>>>(x, xhi, xlo, xq);
        split_bf16_kernel<<<(wq + 255) / 256, 256, 0, stream>>>(w, whi, wlo, wq);
        screened_linear_async<<<grid, dim3(256), 0, stream>>>(xhi, xlo, whi, wlo, bias, out);
    } else {
        screened_linear_fallback<<<grid, dim3(256), 0, stream>>>(x, w, bias, out);
    }
}
